// MaskedWeight_84069689852525
// MI455X (gfx1250) — compile-verified, matrix operand order checked by reading, not run on hardware
//
#include <hip/hip_runtime.h>

// Problem constants (match the reference)
#define B_   512
#define DIM_ 64
#define IN_  2048
#define OUT_ 2048

#define KC    64                   // K-chunk per LDS stage
#define PADK  72                   // padded LDS row stride (bf16): 144B = 36 dwords
#define TILE_ELEMS (64 * PADK)     // ushorts per array per buffer
#define NKC   (IN_ / KC)

typedef __attribute__((ext_vector_type(16))) __bf16 v16bf;
typedef __attribute__((ext_vector_type(8)))  float  v8f;

// ---- bf16 helpers (bit-level, RNE) ------------------------------------------
__device__ __forceinline__ unsigned short f2bf(float f) {
  unsigned int u = __float_as_uint(f);
  u += 0x7FFFu + ((u >> 16) & 1u);           // round-to-nearest-even
  return (unsigned short)(u >> 16);
}
__device__ __forceinline__ float bf2f(unsigned short h) {
  return __uint_as_float(((unsigned int)h) << 16);
}

// ---- CDNA5 async global->LDS copy (ASYNCcnt path, ISA 15.18.3) --------------
__device__ __forceinline__ void async_ld_b128(unsigned lds_byte_off,
                                              const void* gaddr) {
  asm volatile("global_load_async_to_lds_b128 %0, %1, off"
               :: "v"(lds_byte_off), "v"(gaddr) : "memory");
}
__device__ __forceinline__ void wait_async0() {
  asm volatile("s_wait_asynccnt 0x0" ::: "memory");
}
// Low 32 bits of a generic pointer into LDS are the LDS byte offset (ISA 10.2).
__device__ __forceinline__ unsigned lds_off(const void* p) {
  return (unsigned)(unsigned long long)p;
}

// ---- LDS fragment load: 32B per lane via two ds_load_b128 -------------------
union FragU { uint4 q[2]; v16bf v; };
__device__ __forceinline__ v16bf ldfrag(const unsigned short* p) {
  FragU f;
  f.q[0] = *(const uint4*)(p);
  f.q[1] = *(const uint4*)(p + 8);
  return f.v;
}

// ---- Kernel 1: masked weight -> normalized w_n (split bf16) + diag wpl ------
__global__ __launch_bounds__(256)
void prep_weight(const float* __restrict__ weight,
                 const float* __restrict__ diag_weight,
                 unsigned short* __restrict__ wn_hi,
                 unsigned short* __restrict__ wn_lo,
                 float* __restrict__ blocks) {
  const int o = blockIdx.x;
  const int t = threadIdx.x;
  const int d = o >> 5;                      // block-row index (32-wide blocks)
  const int i0 = t * 8;
  const int ib = i0 >> 5;                    // all 8 elems share one 32-block

  float w[8], raw[8];
  float ssq = 0.f;
#pragma unroll
  for (int j = 0; j < 8; ++j) {
    float wt = weight[o * IN_ + i0 + j];
    raw[j] = wt;
    float v = (ib < d) ? wt : ((ib == d) ? __expf(wt) : 0.f);
    w[j] = v;
    ssq += v * v;
  }

  __shared__ float red[256];
  red[t] = ssq;
  __syncthreads();
  for (int s = 128; s > 0; s >>= 1) {
    if (t < s) red[t] += red[t + s];
    __syncthreads();
  }
  const float wsq   = red[0];
  const float dw    = diag_weight[o];
  const float scale = __expf(dw) * rsqrtf(wsq);

#pragma unroll
  for (int j = 0; j < 8; ++j) {
    float wn = w[j] * scale;
    unsigned short hi = f2bf(wn);
    wn_hi[o * IN_ + i0 + j] = hi;
    wn_lo[o * IN_ + i0 + j] = f2bf(wn - bf2f(hi));
  }

  if (ib == d) {                              // diagonal block of wpl
    const float lg = 0.5f * __logf(wsq);
#pragma unroll
    for (int j = 0; j < 8; ++j) {
      int ii = i0 + j - d * 32;
      blocks[o * 32 + ii] = dw + raw[j] - lg;
    }
  }
}

// ---- Kernel 2: split inputs into bf16 hi/lo ---------------------------------
__global__ __launch_bounds__(256)
void prep_inputs(const float* __restrict__ x,
                 unsigned short* __restrict__ x_hi,
                 unsigned short* __restrict__ x_lo) {
  int base = (blockIdx.x * 256 + threadIdx.x) * 8;
#pragma unroll
  for (int j = 0; j < 8; ++j) {
    float v = x[base + j];
    unsigned short hi = f2bf(v);
    x_hi[base + j] = hi;
    x_lo[base + j] = f2bf(v - bf2f(hi));
  }
}

// ---- Kernel 3: out = inputs @ w_n^T + bias (split-bf16 WMMA, async LDS) -----
// WG = 256 threads = 8 waves, tile 64M x 64N; waves 4M x 2N, each 16M x 32N.
// K chunk 64, double-buffered LDS: DMA of chunk kc+1 overlaps 12 WMMAs of kc.
// One barrier + one s_wait_asynccnt per chunk.
__global__ __launch_bounds__(256)
void gemm_wmma(const unsigned short* __restrict__ x_hi,
               const unsigned short* __restrict__ x_lo,
               const unsigned short* __restrict__ wn_hi,
               const unsigned short* __restrict__ wn_lo,
               const float* __restrict__ bias,
               float* __restrict__ out) {
  __shared__ unsigned short As_hi[2 * TILE_ELEMS], As_lo[2 * TILE_ELEMS];
  __shared__ unsigned short Bs_hi[2 * TILE_ELEMS], Bs_lo[2 * TILE_ELEMS];

  const int tid  = threadIdx.x;
  const int wave = tid >> 5;
  const int lane = tid & 31;
  const int m_wg = blockIdx.y * 64;
  const int n_wg = blockIdx.x * 64;
  const int wm = wave >> 1;                 // 0..3
  const int wn = wave & 1;                  // 0..1
  const int m0 = m_wg + wm * 16;
  const int n0 = n_wg + wn * 32;

  // Async-copy mapping: thread t -> tile row (t>>2); two 16B segments per row
  // (seg and seg+4) covering the 128B of a 64-k row.
  const int row = tid >> 2;
  const int seg = tid & 3;
  const unsigned dA_hi = lds_off(As_hi) + row * (PADK * 2) + seg * 16;
  const unsigned dA_lo = lds_off(As_lo) + row * (PADK * 2) + seg * 16;
  const unsigned dB_hi = lds_off(Bs_hi) + row * (PADK * 2) + seg * 16;
  const unsigned dB_lo = lds_off(Bs_lo) + row * (PADK * 2) + seg * 16;
  const unsigned short* gA_hi = x_hi  + (size_t)(m_wg + row) * IN_ + seg * 8;
  const unsigned short* gA_lo = x_lo  + (size_t)(m_wg + row) * IN_ + seg * 8;
  const unsigned short* gB_hi = wn_hi + (size_t)(n_wg + row) * IN_ + seg * 8;
  const unsigned short* gB_lo = wn_lo + (size_t)(n_wg + row) * IN_ + seg * 8;

  // Fragment read pointers (lane&15 = row in tile, lane>>4 = K half), buffer 0.
  const int r  = lane & 15;
  const int kh = (lane >> 4) * 16;
  const unsigned short* aP_hi  = &As_hi[(wm * 16 + r) * PADK + kh];
  const unsigned short* aP_lo  = &As_lo[(wm * 16 + r) * PADK + kh];
  const unsigned short* b0P_hi = &Bs_hi[(wn * 32 + r) * PADK + kh];
  const unsigned short* b0P_lo = &Bs_lo[(wn * 32 + r) * PADK + kh];
  const unsigned short* b1P_hi = &Bs_hi[(wn * 32 + 16 + r) * PADK + kh];
  const unsigned short* b1P_lo = &Bs_lo[(wn * 32 + 16 + r) * PADK + kh];

  v8f acc0 = {}, acc1 = {};

  auto issue = [&](int kc, int buf) {
    const int off = kc * KC;                       // elements into K
    const unsigned bo = (unsigned)buf * (TILE_ELEMS * 2);  // bytes
    async_ld_b128(dA_hi + bo,      gA_hi + off);
    async_ld_b128(dA_hi + bo + 64, gA_hi + off + 32);
    async_ld_b128(dA_lo + bo,      gA_lo + off);
    async_ld_b128(dA_lo + bo + 64, gA_lo + off + 32);
    async_ld_b128(dB_hi + bo,      gB_hi + off);
    async_ld_b128(dB_hi + bo + 64, gB_hi + off + 32);
    async_ld_b128(dB_lo + bo,      gB_lo + off);
    async_ld_b128(dB_lo + bo + 64, gB_lo + off + 32);
  };

  issue(0, 0);
  for (int kc = 0; kc < NKC; ++kc) {
    const int sel = (kc & 1) * TILE_ELEMS;

    wait_async0();          // only chunk kc's 8 loads are outstanding
    __syncthreads();        // publishes chunk kc; also fences last chunk's reads

    if (kc + 1 < NKC) issue(kc + 1, (kc + 1) & 1);  // DMA overlaps WMMAs below

#pragma unroll
    for (int ks = 0; ks < KC; ks += 32) {
      v16bf a_hi  = ldfrag(aP_hi  + sel + ks);
      v16bf a_lo  = ldfrag(aP_lo  + sel + ks);
      v16bf b0_hi = ldfrag(b0P_hi + sel + ks);
      v16bf b0_lo = ldfrag(b0P_lo + sel + ks);
      v16bf b1_hi = ldfrag(b1P_hi + sel + ks);
      v16bf b1_lo = ldfrag(b1P_lo + sel + ks);

      acc0 = __builtin_amdgcn_wmma_f32_16x16x32_bf16(false, a_hi, false, b0_hi,
                                                     (short)0, acc0, false, false);
      acc0 = __builtin_amdgcn_wmma_f32_16x16x32_bf16(false, a_lo, false, b0_hi,
                                                     (short)0, acc0, false, false);
      acc0 = __builtin_amdgcn_wmma_f32_16x16x32_bf16(false, a_hi, false, b0_lo,
                                                     (short)0, acc0, false, false);
      acc1 = __builtin_amdgcn_wmma_f32_16x16x32_bf16(false, a_hi, false, b1_hi,
                                                     (short)0, acc1, false, false);
      acc1 = __builtin_amdgcn_wmma_f32_16x16x32_bf16(false, a_lo, false, b1_hi,
                                                     (short)0, acc1, false, false);
      acc1 = __builtin_amdgcn_wmma_f32_16x16x32_bf16(false, a_hi, false, b1_lo,
                                                     (short)0, acc1, false, false);
    }
  }

  // C/D layout: lane&15 = N, vgpr rr -> M = rr + 8*(lane>=16)
  const int col0 = n0 + (lane & 15);
  const int rowb = m0 + (lane >> 4) * 8;
  const float bv0 = bias[col0];
  const float bv1 = bias[col0 + 16];
#pragma unroll
  for (int rr = 0; rr < 8; ++rr) {
    out[(size_t)(rowb + rr) * OUT_ + col0]      = acc0[rr] + bv0;
    out[(size_t)(rowb + rr) * OUT_ + col0 + 16] = acc1[rr] + bv1;
  }
}

// ---- Kernel 4: log_jac[b,d,od] = LSE_ii( blocks[d*32+od][ii] + grad[b,d,ii] )
__global__ __launch_bounds__(256)
void logjac(const float* __restrict__ blocks,
            const float* __restrict__ grad,
            float* __restrict__ lj) {
  const int d    = blockIdx.x;
  const int t    = threadIdx.x;
  const int wave = t >> 5;
  const int lane = t & 31;

  __shared__ float Lsh[32 * 33];   // padded stride 33 -> conflict-free
  __shared__ float gsh[8 * 32];

#pragma unroll
  for (int j = 0; j < 4; ++j) {
    int e = t * 4 + j;
    Lsh[(e >> 5) * 33 + (e & 31)] = blocks[d * 1024 + e];
  }
  const int b = blockIdx.y * 8 + wave;
  gsh[wave * 32 + lane] = grad[((size_t)b * 64 + d) * 32 + lane];
  __syncthreads();

  const float* Lr = &Lsh[lane * 33];
  const float* g  = &gsh[wave * 32];
  float m = -3.402823466e38f;
#pragma unroll 4
  for (int ii = 0; ii < 32; ++ii)
    m = fmaxf(m, Lr[ii] + g[ii]);
  float s = 0.f;
#pragma unroll 4
  for (int ii = 0; ii < 32; ++ii)
    s += __expf(Lr[ii] + g[ii] - m);
  lj[((size_t)b * 64 + d) * 32 + lane] = __logf(s) + m;
}

// ---- Launch -----------------------------------------------------------------
extern "C" void kernel_launch(void* const* d_in, const int* in_sizes, int n_in,
                              void* d_out, int out_size, void* d_ws, size_t ws_size,
                              hipStream_t stream) {
  (void)in_sizes; (void)n_in; (void)out_size; (void)ws_size;
  const float* inputs = (const float*)d_in[0];   // [512, 2048]
  const float* grad   = (const float*)d_in[1];   // [512, 64, 32, 1]
  const float* weight = (const float*)d_in[2];   // [2048, 2048]
  const float* diagw  = (const float*)d_in[3];   // [2048, 1]
  const float* bias   = (const float*)d_in[4];   // [2048]

  float* out = (float*)d_out;                    // [512, 2048]
  float* lj  = out + (size_t)B_ * OUT_;          // [512, 64, 32, 1]

  char* ws = (char*)d_ws;                        // ~20.3 MB total
  unsigned short* wn_hi = (unsigned short*)(ws);                  //  8 MB
  unsigned short* wn_lo = (unsigned short*)(ws + (8u  << 20));    //  8 MB
  unsigned short* x_hi  = (unsigned short*)(ws + (16u << 20));    //  2 MB
  unsigned short* x_lo  = (unsigned short*)(ws + (18u << 20));    //  2 MB
  float*          blocks= (float*)        (ws + (20u << 20));     // 256 KB

  prep_weight<<<OUT_, 256, 0, stream>>>(weight, diagw, wn_hi, wn_lo, blocks);
  prep_inputs<<<(B_ * IN_) / (256 * 8), 256, 0, stream>>>(inputs, x_hi, x_lo);
  gemm_wmma<<<dim3(OUT_ / 64, B_ / 64), 256, 0, stream>>>(x_hi, x_lo, wn_hi,
                                                          wn_lo, bias, out);
  logjac<<<dim3(DIM_, B_ / 8), 256, 0, stream>>>(blocks, grad, lj);
}
